// GraphAttentionLayer_34634616275017
// MI455X (gfx1250) — matrix operand order repaired and checked by reading, not compiled, not run on hardware
//
#include <hip/hip_runtime.h>
#include <hip/hip_bf16.h>

typedef __attribute__((ext_vector_type(16))) __bf16 v16bf;
typedef __attribute__((ext_vector_type(8)))  __bf16 v8bf;
typedef __attribute__((ext_vector_type(8)))  float  v8f;

#define N_NODES   3072
#define IN_FEAT   256
#define OUT_FEAT  512
#define N_HEADS   8
#define HEAD_F    64
#define NEG_SLOPE 0.2f

// ---------------------------------------------------------------- converts
__global__ void gat_cvt_x(const float* __restrict__ x, __bf16* __restrict__ xb, int n) {
    int i = blockIdx.x * 256 + threadIdx.x;
    if (i < n) xb[i] = (__bf16)x[i];
}

// W is [IN_FEAT, OUT_FEAT] row-major; store transposed bf16: wt[fo][k]
__global__ void gat_cvt_w(const float* __restrict__ w, __bf16* __restrict__ wt) {
    int i = blockIdx.x * 256 + threadIdx.x;
    if (i < IN_FEAT * OUT_FEAT) {
        int k  = i / OUT_FEAT;
        int fo = i - k * OUT_FEAT;
        wt[fo * IN_FEAT + k] = (__bf16)w[i];
    }
}

// ---------------------------------------------------------------- g = x @ W
// One wave computes a 16(i) x 64(fo) tile. Output stored transposed bf16:
// gT[fo][i], so that the aggregation kernel's B tiles are j-contiguous.
__global__ __launch_bounds__(32) void gat_gemm(const __bf16* __restrict__ xb,
                                               const __bf16* __restrict__ wt,
                                               __bf16* __restrict__ gT) {
    const int lane = threadIdx.x & 31;
    const int lo = lane & 15, hi = lane >> 4;
    const int ibase  = blockIdx.x * 16;
    const int fobase = blockIdx.y * 64;

    v8f acc[4] = {};
    const __bf16* arow = xb + (ibase + lo) * IN_FEAT;

    for (int kb = 0; kb < IN_FEAT; kb += 32) {
        // A 16x32 bf16: lane lo row, K = hi*16 .. hi*16+15 contiguous
        v8bf c0 = *(const v8bf*)(arow + kb + hi * 16);
        v8bf c1 = *(const v8bf*)(arow + kb + hi * 16 + 8);
        v16bf a = __builtin_shufflevector(c0, c1,
            0,1,2,3,4,5,6,7,8,9,10,11,12,13,14,15);
        #pragma unroll
        for (int fc = 0; fc < 4; ++fc) {
            // B 32x16 bf16: lane col = lo, K = hi*16 .. hi*16+15 contiguous
            const __bf16* bp = wt + (fobase + fc * 16 + lo) * IN_FEAT + kb + hi * 16;
            v16bf b = *(const v16bf*)bp;
            acc[fc] = __builtin_amdgcn_wmma_f32_16x16x32_bf16(
                false, a, false, b, (short)0, acc[fc], false, false);
        }
    }
    #pragma unroll
    for (int fc = 0; fc < 4; ++fc) {
        const int fo = fobase + fc * 16 + lo;   // D: N = lane&15
        #pragma unroll
        for (int v = 0; v < 8; ++v) {
            const int ir = ibase + v + 8 * hi;  // D: M = vgpr + 8*(lane>=16)
            gT[fo * N_NODES + ir] = (__bf16)acc[fc][v];
        }
    }
}

// ---------------------------------------------------------------- scores
__global__ void gat_scores(const __bf16* __restrict__ gT, const float* __restrict__ aw,
                           float* __restrict__ slT, float* __restrict__ srT,
                           float* __restrict__ sr_row) {
    int idx = blockIdx.x * 256 + threadIdx.x;
    if (idx >= N_NODES * N_HEADS) return;
    int i = idx >> 3, h = idx & 7;
    float sl = 0.f, sr = 0.f;
    for (int f = 0; f < HEAD_F; ++f) {
        float v = (float)gT[(h * HEAD_F + f) * N_NODES + i];
        sl += v * aw[f];
        sr += v * aw[HEAD_F + f];
    }
    slT[h * N_NODES + i]    = sl;
    srT[h * N_NODES + i]    = sr;
    sr_row[i * N_HEADS + h] = sr;
}

// ---------------------------------------------------------------- row max
// m[i,h] = max( any(adj[i,:]==0) ? 1 : -inf,
//               lrelu(s_l[i,h] + max_{j: adj!=0} s_r[j,h]) )   (lrelu monotonic)
__global__ __launch_bounds__(256) void gat_rowmax(const int* __restrict__ adj,
                                                  const float* __restrict__ sr_row,
                                                  const float* __restrict__ slT,
                                                  float* __restrict__ mT) {
    __shared__ float sm[256 * 8];
    __shared__ int   sz[256];
    const int i = blockIdx.x, tid = threadIdx.x;
    float mx[8];
    #pragma unroll
    for (int h = 0; h < 8; ++h) mx[h] = -3.0e38f;
    int anyz = 0;
    const int* arow = adj + (long)i * N_NODES;
    for (int j = tid; j < N_NODES; j += 256) {
        if (arow[j] == 0) { anyz = 1; }
        else {
            const float4* p = (const float4*)(sr_row + j * 8);
            float4 v0 = p[0], v1 = p[1];
            mx[0] = fmaxf(mx[0], v0.x); mx[1] = fmaxf(mx[1], v0.y);
            mx[2] = fmaxf(mx[2], v0.z); mx[3] = fmaxf(mx[3], v0.w);
            mx[4] = fmaxf(mx[4], v1.x); mx[5] = fmaxf(mx[5], v1.y);
            mx[6] = fmaxf(mx[6], v1.z); mx[7] = fmaxf(mx[7], v1.w);
        }
    }
    #pragma unroll
    for (int h = 0; h < 8; ++h) sm[tid * 8 + h] = mx[h];
    sz[tid] = anyz;
    __syncthreads();
    for (int s = 128; s > 0; s >>= 1) {
        if (tid < s) {
            #pragma unroll
            for (int h = 0; h < 8; ++h)
                sm[tid * 8 + h] = fmaxf(sm[tid * 8 + h], sm[(tid + s) * 8 + h]);
            sz[tid] |= sz[tid + s];
        }
        __syncthreads();
    }
    if (tid < 8) {
        const int h = tid;
        float e = slT[h * N_NODES + i] + sm[h];
        e = e > 0.f ? e : NEG_SLOPE * e;
        float m = sz[0] ? fmaxf(1.0f, e) : e;
        mT[h * N_NODES + i] = m;
    }
}

// ---------------------------------------------------------------- aggregate
// One wave: 16 rows x one head. Streams j in steps of 32, builds bf16 P tile
// in A layout, WMMA-accumulates out += P @ G and the softmax denominator.
__global__ __launch_bounds__(32) void gat_aggregate(const int* __restrict__ adj,
                                                    const __bf16* __restrict__ gT,
                                                    const float* __restrict__ slT,
                                                    const float* __restrict__ srT,
                                                    const float* __restrict__ mT,
                                                    float* __restrict__ out) {
    const int lane = threadIdx.x & 31;
    const int lo = lane & 15, hi = lane >> 4;
    const int h = blockIdx.y;
    const int ibase = blockIdx.x * 16;
    const int i_lane = ibase + lo;

    const float sl = slT[h * N_NODES + i_lane];
    const float m  = mT [h * N_NODES + i_lane];
    const float c0 = __expf(1.0f - m);          // exp(e-m) for masked entries (e==1)

    const int*   arow = adj + (long)i_lane * N_NODES;
    const float* srh  = srT + h * N_NODES;

    v8f acc[4] = {};
    float ssum = 0.f;

    for (int jb = 0; jb < N_NODES; jb += 32) {
        const int j0 = jb + hi * 16;        // A elems 0..7  -> K = hi*16 + t
        const int j1 = jb + hi * 16 + 8;    // A elems 8..15 -> K = hi*16 + 8 + t
        int4 a00 = *(const int4*)(arow + j0);
        int4 a01 = *(const int4*)(arow + j0 + 4);
        int4 a10 = *(const int4*)(arow + j1);
        int4 a11 = *(const int4*)(arow + j1 + 4);
        float4 s00 = *(const float4*)(srh + j0);
        float4 s01 = *(const float4*)(srh + j0 + 4);
        float4 s10 = *(const float4*)(srh + j1);
        float4 s11 = *(const float4*)(srh + j1 + 4);
        if (jb + 32 < N_NODES)
            __builtin_prefetch(arow + j0 + 32, 0, 0);   // global_prefetch_b8

        v16bf a;
        #define GAT_P(az, sv, slot) do {                                   \
            float e_ = sl + (sv);                                          \
            e_ = e_ > 0.f ? e_ : NEG_SLOPE * e_;                           \
            float p_ = (az) ? c0 : __expf(e_ - m);                         \
            ssum += p_;                                                    \
            a[slot] = (__bf16)p_;                                          \
        } while (0)
        GAT_P(a00.x == 0, s00.x, 0);  GAT_P(a00.y == 0, s00.y, 1);
        GAT_P(a00.z == 0, s00.z, 2);  GAT_P(a00.w == 0, s00.w, 3);
        GAT_P(a01.x == 0, s01.x, 4);  GAT_P(a01.y == 0, s01.y, 5);
        GAT_P(a01.z == 0, s01.z, 6);  GAT_P(a01.w == 0, s01.w, 7);
        GAT_P(a10.x == 0, s10.x, 8);  GAT_P(a10.y == 0, s10.y, 9);
        GAT_P(a10.z == 0, s10.z, 10); GAT_P(a10.w == 0, s10.w, 11);
        GAT_P(a11.x == 0, s11.x, 12); GAT_P(a11.y == 0, s11.y, 13);
        GAT_P(a11.z == 0, s11.z, 14); GAT_P(a11.w == 0, s11.w, 15);
        #undef GAT_P

        #pragma unroll
        for (int fc = 0; fc < 4; ++fc) {
            const __bf16* bp = gT + (h * HEAD_F + fc * 16 + lo) * N_NODES + jb + hi * 16;
            v16bf b = *(const v16bf*)bp;
            acc[fc] = __builtin_amdgcn_wmma_f32_16x16x32_bf16(
                false, a, false, b, (short)0, acc[fc], false, false);
        }
    }

    // Each lane holds a partial denominator for its row (lanes l and l+16 split j).
    ssum += __shfl_xor(ssum, 16, 32);        // full row sum, row r at lanes r, r+16
    float inv[8];
    #pragma unroll
    for (int v = 0; v < 8; ++v)
        inv[v] = 1.0f / __shfl(ssum, v + 8 * hi, 32);   // match D-row layout

    #pragma unroll
    for (int fc = 0; fc < 4; ++fc) {
        const int col = h * HEAD_F + fc * 16 + lo;
        #pragma unroll
        for (int v = 0; v < 8; ++v)
            out[(long)(ibase + v + 8 * hi) * OUT_FEAT + col] = acc[fc][v] * inv[v];
    }
}

// ---------------------------------------------------------------- launcher
extern "C" void kernel_launch(void* const* d_in, const int* in_sizes, int n_in,
                              void* d_out, int out_size, void* d_ws, size_t ws_size,
                              hipStream_t stream) {
    const float* x   = (const float*)d_in[0];
    const int*   adj = (const int*)  d_in[1];
    const float* W   = (const float*)d_in[2];
    const float* aw  = (const float*)d_in[3];
    float* out = (float*)d_out;

    char* ws = (char*)d_ws;
    __bf16* xb     = (__bf16*)(ws);                 // 3072*256*2  = 1,572,864
    __bf16* wt     = (__bf16*)(ws + 1572864);       // 512*256*2   =   262,144
    __bf16* gT     = (__bf16*)(ws + 1835008);       // 512*3072*2  = 3,145,728
    float*  slT    = (float*) (ws + 4980736);       // 8*3072*4    =    98,304
    float*  srT    = (float*) (ws + 5079040);       //                  98,304
    float*  mT     = (float*) (ws + 5177344);       //                  98,304
    float*  sr_row = (float*) (ws + 5275648);       //                  98,304

    gat_cvt_x<<<(N_NODES * IN_FEAT + 255) / 256, 256, 0, stream>>>(x, xb, N_NODES * IN_FEAT);
    gat_cvt_w<<<(IN_FEAT * OUT_FEAT + 255) / 256, 256, 0, stream>>>(W, wt);

    dim3 gg(N_NODES / 16, OUT_FEAT / 64);
    gat_gemm<<<gg, 32, 0, stream>>>(xb, wt, gT);

    gat_scores<<<(N_NODES * N_HEADS + 255) / 256, 256, 0, stream>>>(gT, aw, slT, srT, sr_row);
    gat_rowmax<<<N_NODES, 256, 0, stream>>>(adj, sr_row, slT, mT);

    dim3 ga(N_NODES / 16, N_HEADS);
    gat_aggregate<<<ga, 32, 0, stream>>>(adj, gT, slT, srT, mT, out);
}
